// SR_GNN_Model_34643206209545
// MI455X (gfx1250) — compile-verified, matrix operand order checked by reading, not run on hardware
//
#include <hip/hip_runtime.h>
#include <hip/hip_bf16.h>
#include <math.h>

// ---------------------------------------------------------------------------
// SR-GNN session kernel for MI455X (gfx1250, wave32, WMMA bf16->f32,
// async global->LDS embedding gather). One workgroup (128 thr / 4 waves)
// per session b.
// ---------------------------------------------------------------------------

typedef __attribute__((ext_vector_type(16))) __bf16    v16bf;
typedef __attribute__((ext_vector_type(8)))  __bf16    v8bf;
typedef __attribute__((ext_vector_type(8)))  float     v8f;
typedef __attribute__((ext_vector_type(4)))  unsigned  v4u;

#define NB   4096   // B
#define NT   50     // T
#define ND   64     // D

struct __align__(16) Smem {
  float  h[64][64];      // f32 gathered embeddings (async DMA target, 16 KB)
  __bf16 hb[64][64];     // bf16 shadow of h, rows >= len zeroed (8 KB)
  float  gnn[64][64];    // GRU output f32 (16 KB)
  __bf16 gnnb[64][64];   // bf16 shadow of gnn (8 KB)
  __bf16 magg[64][64];   // m_agg in bf16 (8 KB)
  int    rows[64];       // gathered item ids
  float  g2[64];
  float  scores[64];
  float  attn[64];
  float  hlast[64];
  float  hglob[64];
  float  hsess[64];
  float  red[4];
};

__device__ __forceinline__ v8f wmma_bf16(v16bf a, v16bf b, v8f c) {
  // D = A(16x32 bf16) * B(32x16 bf16) + C(16x16 f32)
  return __builtin_amdgcn_wmma_f32_16x16x32_bf16(false, a, false, b,
                                                 (short)0, c, false, false);
}

// Pack 8+8 consecutive f32 (two 32B groups, 16B-vectorized) into a bf16 tile half.
__device__ __forceinline__ v16bf pack16(float4 a0, float4 a1, float4 b0, float4 b1) {
  v16bf r;
  r[0]  = (__bf16)a0.x; r[1]  = (__bf16)a0.y; r[2]  = (__bf16)a0.z; r[3]  = (__bf16)a0.w;
  r[4]  = (__bf16)a1.x; r[5]  = (__bf16)a1.y; r[6]  = (__bf16)a1.z; r[7]  = (__bf16)a1.w;
  r[8]  = (__bf16)b0.x; r[9]  = (__bf16)b0.y; r[10] = (__bf16)b0.z; r[11] = (__bf16)b0.w;
  r[12] = (__bf16)b1.x; r[13] = (__bf16)b1.y; r[14] = (__bf16)b1.z; r[15] = (__bf16)b1.w;
  return r;
}

// A tile (16x32 bf16) from a bf16 row-major LDS matrix.
// ISA layout: lane&15 = row M, lane>>4 selects K-half (adds 8),
// element e: k = k0 + (e>=8 ? 16 : 0) + khalf*8 + (e&7).
__device__ __forceinline__ v16bf ldA_bf16(const __bf16* X, int ld, int m0, int k0, int lane) {
  int row = m0 + (lane & 15);
  int kh  = (lane >> 4) << 3;
  const v8bf* p = (const v8bf*)(X + row * ld + k0 + kh);
  v8bf lo = p[0], hi = p[2];
  v16bf r;
#pragma unroll
  for (int i = 0; i < 8; ++i) { r[i] = lo[i]; r[8 + i] = hi[i]; }
  return r;
}

// B tile (32x16) for X @ W^T from global f32: B[k][n] = W[n][k]; ld = in-dim.
__device__ __forceinline__ v16bf ldB_w(const float* W, int ld, int n0, int k0, int lane) {
  int n  = n0 + (lane & 15);
  int kh = (lane >> 4) << 3;
  const float4* p = (const float4*)(W + n * ld + k0 + kh);
  return pack16(p[0], p[1], p[4], p[5]);
}

// A tile of Mcat = concat([m_in, m_out], -1) from bf16 shadow hb, shifted rows:
//   k <  64 : m_in[t][k]  = (1<=t<len) ? h[t-1][k]    : 0
//   k >= 64 : m_out[t][k] = (t+1<len)  ? h[t+1][k-64] : 0
// Branchless: clamp row, unconditional b128 load, zero by bitwise cndmask.
__device__ __forceinline__ v16bf ldA_mcat(const __bf16 (*hb)[64], int len, int m0, int k0, int lane) {
  int t  = m0 + (lane & 15);
  int kh = (lane >> 4) << 3;
  v16bf r;
#pragma unroll
  for (int g = 0; g < 2; ++g) {
    int k = k0 + g * 16 + kh;          // k0 in {0,32,64,96}, kh in {0,8}: group stays in one half
    bool lo = (k < 64);
    int srow = lo ? (t - 1) : (t + 1);
    bool val = lo ? ((t >= 1) && (t < len)) : ((t + 1) < len);
    int sr = min(max(srow, 0), 63);
    int kk = lo ? k : (k - 64);
    v8bf v = *(const v8bf*)&hb[sr][kk];
    v4u  u = __builtin_bit_cast(v4u, v);
    v4u  z = {0u, 0u, 0u, 0u};
    u = val ? u : z;
    v = __builtin_bit_cast(v8bf, u);
#pragma unroll
    for (int i = 0; i < 8; ++i) r[g * 8 + i] = v[i];
  }
  return r;
}

__device__ __forceinline__ float sigm(float x) { return 1.0f / (1.0f + __expf(-x)); }

__global__ void __launch_bounds__(128)
srgnn_session_kernel(const int* __restrict__ hist_mat, const int* __restrict__ hist_len,
                     const int* __restrict__ items,    const float* __restrict__ emb,
                     const float* __restrict__ WH_w,   const float* __restrict__ WH_b,
                     const float* __restrict__ w_ih,   const float* __restrict__ w_hh,
                     const float* __restrict__ b_ih,   const float* __restrict__ b_hh,
                     const float* __restrict__ W1_w,   const float* __restrict__ W1_b,
                     const float* __restrict__ W2_w,   const float* __restrict__ W2_b,
                     const float* __restrict__ q_vec,  const float* __restrict__ b_attn,
                     const float* __restrict__ W3_w,   const float* __restrict__ W3_b,
                     float* __restrict__ out)
{
  __shared__ Smem sm;
  const int b    = blockIdx.x;
  const int tid  = threadIdx.x;
  const int lane = tid & 31;
  const int wave = tid >> 5;
  const int len  = hist_len[b];

  // ---- stage row ids, zero hb (padded rows feed the WMMA A operands) ----
  if (tid < NT) sm.rows[tid] = hist_mat[b * NT + tid];
  {
    v4u z = {0u, 0u, 0u, 0u};
    for (int e = tid; e < 64 * 8; e += 128) ((v4u*)sm.hb)[e] = z;
  }
  __syncthreads();

  // ---- async gather: DMA 16B chunks of the embedding rows straight to LDS ----
  // GVS mode: 64-bit SGPR base (emb) + per-lane 32-bit byte offset. Tracked by
  // ASYNCcnt; no VGPR staging, no loadcnt stall at the write site.
  {
    const unsigned ldsBase = (unsigned)(uintptr_t)(&sm.h[0][0]);
    for (int e = tid; e < len * 16; e += 128) {
      int t = e >> 4, c = e & 15;
      unsigned lds_off = ldsBase + (unsigned)(t * 256 + c * 16);
      int goff = sm.rows[t] * 256 + c * 16;       // <= 1M*256B, fits in i32
      asm volatile("global_load_async_to_lds_b128 %0, %1, %2 offset:0"
                   :: "v"(lds_off), "v"(goff), "s"(emb)
                   : "memory");
    }
    asm volatile("s_wait_asynccnt 0" ::: "memory");
  }
  __syncthreads();

  // ---- build bf16 shadow of h ----
  for (int e = tid; e < len * 16; e += 128) {
    int t = e >> 4, c = e & 15;
    float4 v = ((const float4*)&sm.h[t][0])[c];
    __bf16* hb = &sm.hb[t][c * 4];
    hb[0] = (__bf16)v.x; hb[1] = (__bf16)v.y; hb[2] = (__bf16)v.z; hb[3] = (__bf16)v.w;
  }
  __syncthreads();

  // ---- m_agg = Mcat @ WH^T + WH_b   (wave owns n0 = wave*16) ----
  {
    const int n0  = wave * 16;
    const int col = n0 + (lane & 15);
    const float bias = WH_b[col];
    v16bf BW[4];
#pragma unroll
    for (int kk = 0; kk < 4; ++kk) BW[kk] = ldB_w(WH_w, 128, n0, kk * 32, lane);
    for (int m0 = 0; m0 < 64; m0 += 16) {
      v8f acc = {};
#pragma unroll
      for (int kk = 0; kk < 4; ++kk) {
        v16bf a = ldA_mcat(sm.hb, len, m0, kk * 32, lane);
        acc = wmma_bf16(a, BW[kk], acc);
      }
      const int rbase = m0 + ((lane >> 4) << 3);
#pragma unroll
      for (int i = 0; i < 8; ++i)
        sm.magg[rbase + i][col] = (__bf16)(acc[i] + bias);
    }
  }
  __syncthreads();

  // ---- fused GRUCell: gx = magg @ w_ih^T, gh = h @ w_hh^T, gates in regs ----
  {
    const int d0  = wave * 16;
    const int col = d0 + (lane & 15);
    const float bir  = b_ih[col],        bhr = b_hh[col];
    const float biz  = b_ih[64 + col],   bhz = b_hh[64 + col];
    const float bin_ = b_ih[128 + col],  bhn = b_hh[128 + col];
    v16bf Bih[3][2], Bhh[3][2];
#pragma unroll
    for (int g = 0; g < 3; ++g)
#pragma unroll
      for (int kk = 0; kk < 2; ++kk) {
        Bih[g][kk] = ldB_w(w_ih, 64, g * 64 + d0, kk * 32, lane);
        Bhh[g][kk] = ldB_w(w_hh, 64, g * 64 + d0, kk * 32, lane);
      }
    for (int m0 = 0; m0 < 64; m0 += 16) {
      v8f xr = {}, xz = {}, xn = {}, hr = {}, hz = {}, hn = {};
#pragma unroll
      for (int kk = 0; kk < 2; ++kk) {
        v16bf ax = ldA_bf16(&sm.magg[0][0], 64, m0, kk * 32, lane);
        v16bf ah = ldA_bf16(&sm.hb[0][0],   64, m0, kk * 32, lane);
        xr = wmma_bf16(ax, Bih[0][kk], xr);
        xz = wmma_bf16(ax, Bih[1][kk], xz);
        xn = wmma_bf16(ax, Bih[2][kk], xn);
        hr = wmma_bf16(ah, Bhh[0][kk], hr);
        hz = wmma_bf16(ah, Bhh[1][kk], hz);
        hn = wmma_bf16(ah, Bhh[2][kk], hn);
      }
      const int rbase = m0 + ((lane >> 4) << 3);
#pragma unroll
      for (int i = 0; i < 8; ++i) {
        int row = rbase + i;
        float r  = sigm(xr[i] + bir + hr[i] + bhr);
        float z  = sigm(xz[i] + biz + hz[i] + bhz);
        float n  = tanhf(xn[i] + bin_ + r * (hn[i] + bhn));
        float hv = sm.h[row][col];
        float hnw = (1.0f - z) * n + z * hv;
        float g  = (row < len) ? hnw : 0.0f;   // invalid rows get ~0 attn
        sm.gnn[row][col]  = g;
        sm.gnnb[row][col] = (__bf16)g;
      }
    }
  }
  __syncthreads();

  // ---- g2 = h_last @ W2^T + W2_b + b_attn ; init scores ----
  const int last = len - 1;
  if (tid < 64) {
    float s = W2_b[tid] + b_attn[tid];
    const float* hl = &sm.gnn[last][0];
    for (int k = 0; k < 64; ++k) s += hl[k] * W2_w[tid * 64 + k];
    sm.g2[tid]    = s;
    sm.hlast[tid] = hl[tid];
    sm.scores[tid] = 0.0f;
  }
  __syncthreads();

  // ---- gate = sigmoid(gnn @ W1^T + W1_b + g2); scores[t] = <gate, q> ----
  {
    const int d0  = wave * 16;
    const int col = d0 + (lane & 15);
    const float add = W1_b[col] + sm.g2[col];
    const float qv  = q_vec[col];
    v16bf BW[2];
#pragma unroll
    for (int kk = 0; kk < 2; ++kk) BW[kk] = ldB_w(W1_w, 64, d0, kk * 32, lane);
    for (int m0 = 0; m0 < 64; m0 += 16) {
      v8f acc = {};
#pragma unroll
      for (int kk = 0; kk < 2; ++kk) {
        v16bf a = ldA_bf16(&sm.gnnb[0][0], 64, m0, kk * 32, lane);
        acc = wmma_bf16(a, BW[kk], acc);
      }
      const int rbase = m0 + ((lane >> 4) << 3);
#pragma unroll
      for (int i = 0; i < 8; ++i)
        atomicAdd(&sm.scores[rbase + i], sigm(acc[i] + add) * qv);  // ds_add_f32
    }
  }
  __syncthreads();

  // ---- softmax over valid t ----
  if (tid == 0) {
    float m = -3.0e38f;
    for (int t = 0; t < len; ++t) m = fmaxf(m, sm.scores[t]);
    float ssum = 0.0f;
    for (int t = 0; t < len; ++t) {
      float e = __expf(sm.scores[t] - m);
      sm.attn[t] = e; ssum += e;
    }
    sm.red[0] = ssum;
  }
  __syncthreads();
  if (tid < 64) {
    float inv = 1.0f / sm.red[0];
    sm.attn[tid] = (tid < len) ? sm.attn[tid] * inv : 0.0f;
  }
  __syncthreads();

  // ---- h_global = sum_t attn[t] * gnn[t] ----
  if (tid < 64) {
    float g = 0.0f;
    for (int t = 0; t < len; ++t) g += sm.attn[t] * sm.gnn[t][tid];
    sm.hglob[tid] = g;
  }
  __syncthreads();

  // ---- h_session = [h_last, h_global] @ W3^T + b ; item emb fetch (2nd out) ----
  if (tid < 64) {
    float v = W3_b[tid];
    const float* w3 = W3_w + tid * 128;
    for (int k = 0; k < 64; ++k) v += sm.hlast[k] * w3[k] + sm.hglob[k] * w3[64 + k];
    sm.hsess[tid] = v;
  } else {
    int d = tid - 64;                      // attn[] is dead now: reuse for item emb
    sm.attn[d] = emb[(size_t)items[b] * ND + d];
  }
  __syncthreads();

  // ---- L2 norms and store both outputs ----
  if (tid == 0) {
    float s1 = 0.0f, s2 = 0.0f;
    for (int k = 0; k < 64; ++k) { s1 += sm.hsess[k] * sm.hsess[k]; s2 += sm.attn[k] * sm.attn[k]; }
    sm.red[0] = fmaxf(sqrtf(s1), 1e-12f);
    sm.red[1] = fmaxf(sqrtf(s2), 1e-12f);
  }
  __syncthreads();
  if (tid < 64) {
    out[(size_t)b * ND + tid] = sm.hsess[tid] / sm.red[0];
    out[(size_t)NB * ND + (size_t)b * ND + tid] = sm.attn[tid] / sm.red[1];
  }
}

extern "C" void kernel_launch(void* const* d_in, const int* in_sizes, int n_in,
                              void* d_out, int out_size, void* d_ws, size_t ws_size,
                              hipStream_t stream) {
  (void)in_sizes; (void)n_in; (void)d_ws; (void)ws_size; (void)out_size;
  srgnn_session_kernel<<<NB, 128, 0, stream>>>(
      (const int*)d_in[0],   // item_history_matrix
      (const int*)d_in[1],   // item_history_len
      (const int*)d_in[2],   // items
      (const float*)d_in[3], // item_emb_w
      (const float*)d_in[4], (const float*)d_in[5],   // W_H_w, W_H_b
      (const float*)d_in[6], (const float*)d_in[7],   // gru_w_ih, gru_w_hh
      (const float*)d_in[8], (const float*)d_in[9],   // gru_b_ih, gru_b_hh
      (const float*)d_in[10], (const float*)d_in[11], // W1_w, W1_b
      (const float*)d_in[12], (const float*)d_in[13], // W2_w, W2_b
      (const float*)d_in[14], (const float*)d_in[15], // q_vec, b_attn
      (const float*)d_in[16], (const float*)d_in[17], // W3_w, W3_b
      (float*)d_out);
}
